// GIN_edge_12919261627157
// MI455X (gfx1250) — matrix operand order repaired and checked by reading, not compiled, not run on hardware
//
#include <hip/hip_runtime.h>
#include <hip/hip_bf16.h>

typedef __attribute__((ext_vector_type(16))) _Float16 v16h;
typedef __attribute__((ext_vector_type(8)))  float    v8f;

#define TILES_PER_BLOCK 4

// ---------------------------------------------------------------------------
// CDNA5 WMMA wave32 operand layouts:
//   A (16x32 f16): lane m = L&15; elems 0..7  -> K = 8*hi + i
//                                 elems 8..15 -> K = 16 + 8*hi + i
//   B (32x16 f16): lane n = L&15; elems 0..7 -> K = 8*hi + i, 8..15 -> K = 16 + 8*hi + i
//   D (16x16 f32): lane n = L&15; vgpr r -> row m = r + 8*hi
// ---------------------------------------------------------------------------

// Dense B tile from row-major [64 x 64] f32 weight; kbase = chunk*32 + 16*hi.
__device__ inline v16h load_B_tile64(const float* __restrict__ W, int kbase, int n) {
  v16h b;
#pragma unroll
  for (int i = 0; i < 16; ++i)
    b[i] = (_Float16)W[((kbase & ~31) + 8 * ((kbase >> 4) & 1) + (i & 7) + 16 * (i >> 3)) * 64 + n];
  return b;
}

// B tile for the K=7 first layer (branchless zero-padding: unconditional
// in-bounds load, then value select on hi -- avoids per-element exec masking).
__device__ inline v16h load_B_tile7(const float* __restrict__ W, int hi, int n) {
  v16h b;
#pragma unroll
  for (int i = 0; i < 16; ++i) {
    float w = 0.0f;
    if (i < 7) {                      // compile-time
      float t = W[i * 64 + n];        // always in-bounds, unconditional load
      w = (hi == 0) ? t : 0.0f;       // v_cndmask, no branch
    }
    b[i] = (_Float16)w;
  }
  return b;
}

// A tile from an LDS-staged f16 [16 x 64] activation tile; kb = chunk*32 + 8*hi.
__device__ inline v16h load_A_lds(const _Float16* __restrict__ sA, int m, int kb) {
  v16h a;
  const _Float16* p = sA + m * 64 + kb;
#pragma unroll
  for (int i = 0; i < 8; ++i) { a[i] = p[i]; a[8 + i] = p[16 + i]; }
  return a;
}

// ---------------------------------------------------------------------------
// Edge kernels (bandwidth/atomic bound)
// ---------------------------------------------------------------------------
__global__ void edge1_kernel(const float* __restrict__ ea,
                             const float* __restrict__ We,   // [4,7]
                             const float* __restrict__ be,   // [7]
                             const float* __restrict__ x,    // [N,7]
                             const int*   __restrict__ ei,   // [2,E]
                             float* __restrict__ agg1,       // [N,8]
                             int E) {
  int e = blockIdx.x * blockDim.x + threadIdx.x;
  if (e >= E) return;
  int src = ei[e];
  int dst = ei[E + e];
  float a0 = ea[e * 4 + 0], a1 = ea[e * 4 + 1];
  float a2 = ea[e * 4 + 2], a3 = ea[e * 4 + 3];
#pragma unroll
  for (int j = 0; j < 7; ++j) {
    float v = be[j] + a0 * We[j] + a1 * We[7 + j] + a2 * We[14 + j] + a3 * We[21 + j];
    v += x[src * 7 + j];
    v = fmaxf(v, 0.0f);
    atomicAdd(&agg1[dst * 8 + j], v);
  }
}

// 8 threads per edge, 8 channels each -> 32B contiguous gathers of h1[src].
__global__ void edge2_kernel(const float* __restrict__ ea,
                             const float* __restrict__ We,   // [4,64]
                             const float* __restrict__ be,   // [64]
                             const float* __restrict__ h1,   // [N,64]
                             const int*   __restrict__ ei,
                             float* __restrict__ agg2,       // [N,64]
                             int E) {
  long long gid = (long long)blockIdx.x * blockDim.x + threadIdx.x;
  int e = (int)(gid >> 3);
  if (e >= E) return;
  int c = ((int)gid & 7) * 8;
  int src = ei[e];
  int dst = ei[E + e];
  float a0 = ea[e * 4 + 0], a1 = ea[e * 4 + 1];
  float a2 = ea[e * 4 + 2], a3 = ea[e * 4 + 3];
#pragma unroll
  for (int jj = 0; jj < 8; ++jj) {
    int j = c + jj;
    float v = be[j] + a0 * We[j] + a1 * We[64 + j] + a2 * We[128 + j] + a3 * We[192 + j];
    v += h1[src * 64 + j];
    v = fmaxf(v, 0.0f);
    atomicAdd(&agg2[dst * 64 + j], v);
  }
}

// ---------------------------------------------------------------------------
// Node MLP 1: h1 = relu( relu((x+agg1) @ W1a + b1a) @ W1b + b1b )
// One wave per block; weights live in registers; loop over 4 node tiles.
// ---------------------------------------------------------------------------
__global__ __launch_bounds__(32)
void mlp1_kernel(const float* __restrict__ x, const float* __restrict__ agg1,
                 const float* __restrict__ W1a, const float* __restrict__ b1a,
                 const float* __restrict__ W1b, const float* __restrict__ b1b,
                 float* __restrict__ h1, int nNodes) {
  __shared__ _Float16 sA[16 * 64];
  int lane = threadIdx.x;
  int hi = lane >> 4;
  int m = lane & 15;

  // Register-resident weights (reused across all tiles this wave processes).
  v16h Ba[4], Bb[4][2];
  float biasA[4], biasB[4];
#pragma unroll
  for (int nt = 0; nt < 4; ++nt) {
    int n = nt * 16 + m;
    biasA[nt] = b1a[n];
    biasB[nt] = b1b[n];
    Ba[nt]    = load_B_tile7(W1a, hi, n);
    Bb[nt][0] = load_B_tile64(W1b, 0 + 16 * hi, n);
    Bb[nt][1] = load_B_tile64(W1b, 32 + 16 * hi, n);
  }

  for (int t = 0; t < TILES_PER_BLOCK; ++t) {
    int base = (blockIdx.x * TILES_PER_BLOCK + t) * 16;
    int node = base + m;
    if (node >= nNodes) node = nNodes - 1;  // clamp reads, keep EXEC full

    // A for layer a: K = 7 (zero-padded). Only lanes 0-15 elems 0..6 carry data.
    v16h a0;
#pragma unroll
    for (int i = 0; i < 16; ++i) a0[i] = (_Float16)0.0f;
    if (hi == 0) {
#pragma unroll
      for (int i = 0; i < 7; ++i)
        a0[i] = (_Float16)(x[node * 7 + i] + agg1[node * 8 + i]);
    }

    v8f zero = {};
#pragma unroll
    for (int nt = 0; nt < 4; ++nt) {
      int n = nt * 16 + m;
      v8f acc = __builtin_amdgcn_wmma_f32_16x16x32_f16(false, a0, false, Ba[nt],
                                                       (short)0, zero, false, false);
#pragma unroll
      for (int r = 0; r < 8; ++r) {
        int row = r + 8 * hi;
        sA[row * 64 + n] = (_Float16)fmaxf(acc[r] + biasA[nt], 0.0f);
      }
    }
    __syncthreads();

    v16h ac0 = load_A_lds(sA, m, 0 + 8 * hi);
    v16h ac1 = load_A_lds(sA, m, 32 + 8 * hi);
#pragma unroll
    for (int nt = 0; nt < 4; ++nt) {
      int n = nt * 16 + m;
      v8f acc = {};
      acc = __builtin_amdgcn_wmma_f32_16x16x32_f16(false, ac0, false, Bb[nt][0],
                                                   (short)0, acc, false, false);
      acc = __builtin_amdgcn_wmma_f32_16x16x32_f16(false, ac1, false, Bb[nt][1],
                                                   (short)0, acc, false, false);
#pragma unroll
      for (int r = 0; r < 8; ++r) {
        int row = base + r + 8 * hi;
        if (row < nNodes)
          h1[row * 64 + n] = fmaxf(acc[r] + biasB[nt], 0.0f);
      }
    }
    __syncthreads();  // sA reused next iteration
  }
}

// ---------------------------------------------------------------------------
// Node MLP 2 fused with global mean-pool accumulation:
//   h2 = relu( relu((h1+agg2) @ W2a + b2a) @ W2b + b2b ); pooled[batch[n]] += h2[n]
// ---------------------------------------------------------------------------
__global__ __launch_bounds__(32)
void mlp2_pool_kernel(const float* __restrict__ h1, const float* __restrict__ agg2,
                      const float* __restrict__ W2a, const float* __restrict__ b2a,
                      const float* __restrict__ W2b, const float* __restrict__ b2b,
                      const int* __restrict__ batch,
                      float* __restrict__ pooled, int nNodes) {
  __shared__ _Float16 sA[16 * 64];
  int lane = threadIdx.x;
  int hi = lane >> 4;
  int m = lane & 15;

  v16h Ba[4][2], Bb[4][2];
  float biasA[4], biasB[4];
#pragma unroll
  for (int nt = 0; nt < 4; ++nt) {
    int n = nt * 16 + m;
    biasA[nt] = b2a[n];
    biasB[nt] = b2b[n];
    Ba[nt][0] = load_B_tile64(W2a, 0 + 16 * hi, n);
    Ba[nt][1] = load_B_tile64(W2a, 32 + 16 * hi, n);
    Bb[nt][0] = load_B_tile64(W2b, 0 + 16 * hi, n);
    Bb[nt][1] = load_B_tile64(W2b, 32 + 16 * hi, n);
  }

  for (int t = 0; t < TILES_PER_BLOCK; ++t) {
    int base = (blockIdx.x * TILES_PER_BLOCK + t) * 16;
    int node = base + m;
    if (node >= nNodes) node = nNodes - 1;

    // A chunks from global: hpre = h1 + agg2 (full K = 64).
    v16h ac[2];
#pragma unroll
    for (int c = 0; c < 2; ++c) {
      int kb = c * 32 + 8 * hi;
#pragma unroll
      for (int i = 0; i < 8; ++i) {
        ac[c][i]     = (_Float16)(h1[node * 64 + kb + i]      + agg2[node * 64 + kb + i]);
        ac[c][8 + i] = (_Float16)(h1[node * 64 + kb + 16 + i] + agg2[node * 64 + kb + 16 + i]);
      }
    }

#pragma unroll
    for (int nt = 0; nt < 4; ++nt) {
      int n = nt * 16 + m;
      v8f acc = {};
      acc = __builtin_amdgcn_wmma_f32_16x16x32_f16(false, ac[0], false, Ba[nt][0],
                                                   (short)0, acc, false, false);
      acc = __builtin_amdgcn_wmma_f32_16x16x32_f16(false, ac[1], false, Ba[nt][1],
                                                   (short)0, acc, false, false);
#pragma unroll
      for (int r = 0; r < 8; ++r) {
        int row = r + 8 * hi;
        sA[row * 64 + n] = (_Float16)fmaxf(acc[r] + biasA[nt], 0.0f);
      }
    }
    __syncthreads();

    v16h bc0 = load_A_lds(sA, m, 0 + 8 * hi);
    v16h bc1 = load_A_lds(sA, m, 32 + 8 * hi);

    int g[8];
#pragma unroll
    for (int r = 0; r < 8; ++r) {
      int row = base + r + 8 * hi;
      g[r] = (row < nNodes) ? batch[row] : -1;
    }

#pragma unroll
    for (int nt = 0; nt < 4; ++nt) {
      int n = nt * 16 + m;
      v8f acc = {};
      acc = __builtin_amdgcn_wmma_f32_16x16x32_f16(false, bc0, false, Bb[nt][0],
                                                   (short)0, acc, false, false);
      acc = __builtin_amdgcn_wmma_f32_16x16x32_f16(false, bc1, false, Bb[nt][1],
                                                   (short)0, acc, false, false);
#pragma unroll
      for (int r = 0; r < 8; ++r) {
        float v = fmaxf(acc[r] + biasB[nt], 0.0f);
        if (g[r] >= 0)
          atomicAdd(&pooled[g[r] * 64 + n], v);
      }
    }
    __syncthreads();
  }
}

// ---------------------------------------------------------------------------
// Counts + final FC
// ---------------------------------------------------------------------------
__global__ void count_kernel(const int* __restrict__ batch, float* __restrict__ counts,
                             int nNodes) {
  int n = blockIdx.x * blockDim.x + threadIdx.x;
  if (n < nNodes) atomicAdd(&counts[batch[n]], 1.0f);
}

__global__ void fc_kernel(const float* __restrict__ pooled,
                          const float* __restrict__ counts,
                          const float* __restrict__ Wfc,  // [64,12]
                          const float* __restrict__ bfc,  // [12]
                          float* __restrict__ out, int G) {
  int gidx = blockIdx.x * blockDim.x + threadIdx.x;
  if (gidx >= G) return;
  float inv = 1.0f / fmaxf(counts[gidx], 1.0f);
  float s[12];
#pragma unroll
  for (int j = 0; j < 12; ++j) s[j] = bfc[j];
  for (int k = 0; k < 64; ++k) {
    float p = pooled[gidx * 64 + k] * inv;
#pragma unroll
    for (int j = 0; j < 12; ++j) s[j] += p * Wfc[k * 12 + j];
  }
#pragma unroll
  for (int j = 0; j < 12; ++j) out[gidx * 12 + j] = s[j];
}

// ---------------------------------------------------------------------------
extern "C" void kernel_launch(void* const* d_in, const int* in_sizes, int n_in,
                              void* d_out, int out_size, void* d_ws, size_t ws_size,
                              hipStream_t stream) {
  const float* x    = (const float*)d_in[0];
  const float* ea   = (const float*)d_in[1];
  const float* W_e1 = (const float*)d_in[2];
  const float* b_e1 = (const float*)d_in[3];
  const float* W_e2 = (const float*)d_in[4];
  const float* b_e2 = (const float*)d_in[5];
  const float* W1a  = (const float*)d_in[6];
  const float* b1a  = (const float*)d_in[7];
  const float* W1b  = (const float*)d_in[8];
  const float* b1b  = (const float*)d_in[9];
  const float* W2a  = (const float*)d_in[10];
  const float* b2a  = (const float*)d_in[11];
  const float* W2b  = (const float*)d_in[12];
  const float* b2b  = (const float*)d_in[13];
  const float* Wfc  = (const float*)d_in[14];
  const float* bfc  = (const float*)d_in[15];
  const int*   ei   = (const int*)d_in[16];
  const int*   batch= (const int*)d_in[17];
  float* out = (float*)d_out;

  const int N = in_sizes[0] / 7;
  const int E = in_sizes[1] / 4;
  const int G = out_size / 12;

  // workspace layout (256B aligned)
  char* ws = (char*)d_ws;
  size_t off = 0;
  auto alloc = [&](size_t bytes) { void* p = ws + off; off = (off + bytes + 255) & ~(size_t)255; return p; };
  float* agg1   = (float*)alloc((size_t)N * 8 * 4);
  float* agg2   = (float*)alloc((size_t)N * 64 * 4);
  float* h1     = (float*)alloc((size_t)N * 64 * 4);
  float* pooled = (float*)alloc((size_t)G * 64 * 4);
  float* counts = (float*)alloc((size_t)G * 4);
  (void)ws_size;

  hipMemsetAsync(agg1,   0, (size_t)N * 8 * 4,  stream);
  hipMemsetAsync(agg2,   0, (size_t)N * 64 * 4, stream);
  hipMemsetAsync(pooled, 0, (size_t)G * 64 * 4, stream);
  hipMemsetAsync(counts, 0, (size_t)G * 4,      stream);

  // Layer 1 edge phase
  edge1_kernel<<<(E + 255) / 256, 256, 0, stream>>>(ea, W_e1, b_e1, x, ei, agg1, E);

  // Layer 1 node MLP (WMMA)
  int nTiles = (N + 15) / 16;
  int nBlocks = (nTiles + TILES_PER_BLOCK - 1) / TILES_PER_BLOCK;
  mlp1_kernel<<<nBlocks, 32, 0, stream>>>(x, agg1, W1a, b1a, W1b, b1b, h1, N);

  // Layer 2 edge phase (8 threads per edge)
  long long t2 = (long long)E * 8;
  edge2_kernel<<<(int)((t2 + 255) / 256), 256, 0, stream>>>(ea, W_e2, b_e2, h1, ei, agg2, E);

  // Pool counts
  count_kernel<<<(N + 255) / 256, 256, 0, stream>>>(batch, counts, N);

  // Layer 2 node MLP (WMMA) fused with pooled-sum accumulation
  mlp2_pool_kernel<<<nBlocks, 32, 0, stream>>>(h1, agg2, W2a, b2a, W2b, b2b,
                                               batch, pooled, N);

  // Final classifier with mean normalization
  fc_kernel<<<(G + 127) / 128, 128, 0, stream>>>(pooled, counts, Wfc, bfc, out, G);
}
